// SSMaxEmotionSelfAttention_7215545057733
// MI455X (gfx1250) — matrix operand order repaired and checked by reading, compiled, not run on hardware
//
#include <hip/hip_runtime.h>
#include <hip/hip_bf16.h>
#include <math.h>

// Problem constants (match reference)
constexpr int EE = 512;   // embed
constexpr int NH = 8;     // heads
constexpr int NB = 2;     // batch
constexpr int TT = 2048;  // seq len
constexpr int DD = 64;    // head dim
#define LAMBDA 8.0f
#define GAMMA_ 0.5f

// padded LDS row (floats); pad=4 de-conflicts the 16-row striped stores.
// score stripe: 16*2052*4 = 131328 B ; + 4KB partial-acc scratch = 135424 B
// -> 2 workgroups per 320KB WGP.
constexpr int LROW = TT + 4;
constexpr int PART_OFF = 16 * LROW;             // float offset of scratch
constexpr size_t LDS_BYTES = (size_t)(16 * LROW + 4 * 8 * 32) * sizeof(float);

typedef __attribute__((ext_vector_type(16))) __bf16 v16bf;
typedef __attribute__((ext_vector_type(8)))  __bf16 v8bf;
typedef __attribute__((ext_vector_type(8)))  float  v8f;
typedef __attribute__((ext_vector_type(4)))  float  v4f;

__device__ __forceinline__ v8f wmma_bf16(v16bf a, v16bf b, v8f c) {
  // (neg_a, A, neg_b, B, c_mod, C, reuse_a, reuse_b)
  return __builtin_amdgcn_wmma_f32_16x16x32_bf16(false, a, false, b, (short)0, c,
                                                 false, false);
}

// A-matrix fragment (16x32 bf16): lane L holds row M=L&15.
// lanes 0-15: K = {0..7, 16..23}; lanes 16-31: K = {8..15, 24..31}
__device__ __forceinline__ v16bf load_a16(const __bf16* __restrict__ base,
                                          int row, int ld, int kk, int lane) {
  const __bf16* p = base + (size_t)row * ld + kk + ((lane & 16) >> 1);
  v8bf lo = *(const v8bf*)p;        // K offset +0..7
  v8bf hi = *(const v8bf*)(p + 16); // K offset +16..23
  v16bf r;
#pragma unroll
  for (int i = 0; i < 8; ++i) { r[i] = lo[i]; r[i + 8] = hi[i]; }
  return r;
}

// B-matrix fragment (32x16 bf16), source stored as [N][K] row-major
// (i.e. we compute A * src^T).  lane L holds column N=L&15;
// lanes 0-15: K = {0..7, 16..23}; lanes 16-31: K = {8..15, 24..31}
__device__ __forceinline__ v16bf load_b16_nk(const __bf16* __restrict__ base,
                                             int nrow, int ld, int kk, int lane) {
  const __bf16* p = base + (size_t)nrow * ld + kk + ((lane & 16) >> 1);
  v8bf lo = *(const v8bf*)p;
  v8bf hi = *(const v8bf*)(p + 16);
  v16bf r;
#pragma unroll
  for (int i = 0; i < 8; ++i) { r[i] = lo[i]; r[i + 8] = hi[i]; }
  return r;
}

// ---------------- fp32 -> bf16 convert ----------------
__global__ void cvt_bf16(const float* __restrict__ src, __bf16* __restrict__ dst,
                         int n) {
  int i = blockIdx.x * blockDim.x + threadIdx.x;
  if (i < n) dst[i] = (__bf16)src[i];
}

// ---------------- per-batch bias mean/std (analytic) ----------------
// bias_bt[b,i,j] = e[b,i] * exp(-|i-j|/lam).  Sums factorize:
//   sum  = sum_i e_i * RS1(i),  RS1(i) = sum_j r^|i-j|,  r = exp(-1/lam)
//   sum2 = sum_i e_i^2 * RS2(i), RS2 uses r^2 (geometric closed forms).
__global__ void bias_stats(const float* __restrict__ emo,
                           float* __restrict__ stats) {
  int b = blockIdx.x;
  __shared__ float s1[256], s2[256];
  const float r1 = expf(-1.0f / LAMBDA);
  const float r2 = expf(-2.0f / LAMBDA);
  const float i1 = 1.0f / (1.0f - r1);
  const float i2 = 1.0f / (1.0f - r2);
  float a1 = 0.f, a2 = 0.f;
  for (int i = threadIdx.x; i < TT; i += blockDim.x) {
    float rs1 = (1.0f - expf(-(float)(i + 1) / LAMBDA)) * i1 +
                r1 * (1.0f - expf(-(float)(TT - 1 - i) / LAMBDA)) * i1;
    float rs2 = (1.0f - expf(-2.0f * (float)(i + 1) / LAMBDA)) * i2 +
                r2 * (1.0f - expf(-2.0f * (float)(TT - 1 - i) / LAMBDA)) * i2;
    float e = emo[(size_t)b * TT + i];
    a1 += e * rs1;
    a2 += e * e * rs2;
  }
  s1[threadIdx.x] = a1; s2[threadIdx.x] = a2;
  __syncthreads();
  for (int off = 128; off > 0; off >>= 1) {
    if ((int)threadIdx.x < off) {
      s1[threadIdx.x] += s1[threadIdx.x + off];
      s2[threadIdx.x] += s2[threadIdx.x + off];
    }
    __syncthreads();
  }
  if (threadIdx.x == 0) {
    double N = (double)TT * (double)TT;
    double mean = (double)s1[0] / N;
    double var = ((double)s2[0] - N * mean * mean) / (N - 1.0);
    float sd = sqrtf(fmaxf((float)var, 0.0f));
    stats[b * 2 + 0] = (float)mean;
    stats[b * 2 + 1] = fmaxf(sd, 1e-6f);
  }
}

// ---------------- projection: dst = x @ W^T + bias ----------------
// one wave = one 16x16 output tile.
// vtranspose==0: dst[b,h,t,d] (row-major [T,D] per head)
// vtranspose==1: dst[b,h,d,t] (transposed, so PV B-fragments are contiguous)
__global__ void proj_wmma(const __bf16* __restrict__ Xb, const __bf16* __restrict__ Wb,
                          const float* __restrict__ bias, __bf16* __restrict__ dst,
                          int vtranspose) {
  int lane = threadIdx.x & 31;
  int t0 = blockIdx.x * 16;
  int f0 = blockIdx.y * 16;
  int b  = blockIdx.z;
  const __bf16* X = Xb + (size_t)b * TT * EE;
  v8f acc = {};
  for (int kk = 0; kk < EE; kk += 32) {
    v16bf a = load_a16(X, t0 + (lane & 15), EE, kk, lane);
    v16bf w = load_b16_nk(Wb, f0 + (lane & 15), EE, kk, lane);
    acc = wmma_bf16(a, w, acc);
  }
  int n = lane & 15;
  int mo = (lane >> 4) * 8;
  int f = f0 + n;
  int h = f >> 6, d = f & 63;
  float bv = bias[f];
#pragma unroll
  for (int r = 0; r < 8; ++r) {
    int t = t0 + r + mo;
    float val = acc[r] + bv;
    if (vtranspose)
      dst[(((size_t)b * NH + h) * DD + d) * TT + t] = (__bf16)val;
    else
      dst[(((size_t)b * NH + h) * TT + t) * DD + d] = (__bf16)val;
  }
}

// ---------------- fused attention core ----------------
// One workgroup (8 waves, 256 threads) per (bh, 16-row stripe):
//   phase 1: scores (QK^T * scale + emotion bias) -> LDS stripe [16][TT]
//   phase 2: row softmax in LDS (wave32 shfl reductions), single write of attn to HBM
//   phase 3: Y = P @ V: split-K over wave pairs, P from LDS, V^T from L2-resident
//            global, partial accumulators combined through LDS scratch.
__global__ __launch_bounds__(256) void attn_fused(
    const __bf16* __restrict__ Qb, const __bf16* __restrict__ Kb,
    const __bf16* __restrict__ Vtb, const float* __restrict__ emo,
    const float* __restrict__ stats, const float* __restrict__ s_ptr,
    float* __restrict__ attn, __bf16* __restrict__ Y) {
  extern __shared__ float smem[];   // [16][LROW] + [4][8][32] scratch
  const int tid  = threadIdx.x;
  const int lane = tid & 31;
  const int wid  = tid >> 5;        // 0..7
  const int i0   = blockIdx.x * 16;
  const int bh   = blockIdx.y;
  const int b    = bh >> 3, h = bh & 7;
  const __bf16* Q  = Qb  + (size_t)bh * TT * DD;
  const __bf16* K  = Kb  + (size_t)bh * TT * DD;
  const __bf16* Vt = Vtb + (size_t)bh * DD * TT;
  float* attnrow0 = attn + ((size_t)bh * TT + i0) * TT;

  const float scale  = s_ptr[0] * logf((float)TT) * 0.125f; // s*log(T)/sqrt(D)
  const float mean   = stats[b * 2 + 0];
  const float inv_sd = 1.0f / stats[b * 2 + 1];

  const int n  = lane & 15;
  const int mo = (lane >> 4) * 8;

  // per-block row data hoisted out of the tile loop
  float ev[8];
#pragma unroll
  for (int r = 0; r < 8; ++r) ev[r] = emo[(size_t)b * TT + i0 + r + mo];

  // Q fragments: loaded once, reused for all j tiles of this wave
  v16bf aq0 = load_a16(Q, i0 + n, DD, 0, lane);
  v16bf aq1 = load_a16(Q, i0 + n, DD, 32, lane);

  // ---- phase 1: scores into LDS (8 waves x 16 j-tiles) ----
  for (int jt = wid; jt < TT / 16; jt += 8) {
    int j0 = jt * 16;
    // prefetch next K tile for this wave (emits global_prefetch_b8)
    if (j0 + 128 < TT)
      __builtin_prefetch(K + (size_t)(j0 + 128 + n) * DD, 0, 3);
    v16bf k0 = load_b16_nk(K, j0 + n, DD, 0, lane);
    v16bf k1 = load_b16_nk(K, j0 + n, DD, 32, lane);
    v8f acc = {};
    acc = wmma_bf16(aq0, k0, acc);
    acc = wmma_bf16(aq1, k1, acc);
    int j = j0 + n;
#pragma unroll
    for (int r = 0; r < 8; ++r) {
      int i = i0 + r + mo;
      float kexp = __expf(-fabsf((float)(i - j)) * (1.0f / LAMBDA));
      float bvv  = GAMMA_ * (ev[r] * kexp - mean) * inv_sd;
      smem[(size_t)(r + mo) * LROW + j] = acc[r] * scale + bvv;
    }
  }
  __syncthreads();

  // ---- phase 2: softmax per row (2 rows per wave); write attn to HBM once ----
#pragma unroll
  for (int rr = 0; rr < 2; ++rr) {
    int r = wid * 2 + rr;
    float* rowp = smem + (size_t)r * LROW;
    float m = -INFINITY;
    for (int j = lane; j < TT; j += 32) m = fmaxf(m, rowp[j]);
#pragma unroll
    for (int o = 16; o > 0; o >>= 1) m = fmaxf(m, __shfl_xor(m, o, 32));
    float sum = 0.f;
    for (int j = lane; j < TT; j += 32) {
      float v = __expf(rowp[j] - m);
      rowp[j] = v;
      sum += v;
    }
#pragma unroll
    for (int o = 16; o > 0; o >>= 1) sum += __shfl_xor(sum, o, 32);
    float inv = 1.0f / sum;
    float* grow = attnrow0 + (size_t)r * TT;
    for (int j = lane * 4; j < TT; j += 128) {
      v4f v = *(v4f*)(rowp + j);
      v *= inv;
      *(v4f*)(rowp + j) = v;
      *(v4f*)(grow + j) = v;
    }
  }
  __syncthreads();

  // ---- phase 3: Y = P @ V ----
  // wave w owns d0 = (w&3)*16 and s-half (w>>2); 32 WMMAs per wave.
  const int d0 = (wid & 3) * 16;
  const int sh = wid >> 2;
  const int kb = (lane & 16) >> 1;   // A-fragment K sub-offset (0 or 8)
  v8f accY = {};
  const int s_beg = sh * (TT / 2), s_end = s_beg + TT / 2;
  for (int s = s_beg; s < s_end; s += 32) {
    // prefetch next V stripe (emits global_prefetch_b8)
    if (s + 128 < s_end)
      __builtin_prefetch(Vt + (size_t)(d0 + n) * TT + s + 128, 0, 3);
    const float* pr = smem + (size_t)n * LROW + s + kb;
    v4f a0 = *(const v4f*)pr;
    v4f a1 = *(const v4f*)(pr + 4);
    v4f a2 = *(const v4f*)(pr + 16);
    v4f a3 = *(const v4f*)(pr + 20);
    v16bf a;
#pragma unroll
    for (int i = 0; i < 4; ++i) {
      a[i]      = (__bf16)a0[i];
      a[i + 4]  = (__bf16)a1[i];
      a[i + 8]  = (__bf16)a2[i];
      a[i + 12] = (__bf16)a3[i];
    }
    v16bf vb = load_b16_nk(Vt, d0 + n, TT, s, lane);  // B[s][d] = Vt[d][s]
    accY = wmma_bf16(a, vb, accY);
  }
  // combine split-K partials through LDS scratch
  float* part = smem + PART_OFF;    // [4 d-tiles][8 regs][32 lanes]
  if (wid >= 4) {
#pragma unroll
    for (int r = 0; r < 8; ++r)
      part[(((wid - 4) * 8) + r) * 32 + lane] = accY[r];
  }
  __syncthreads();
  if (wid < 4) {
#pragma unroll
    for (int r = 0; r < 8; ++r)
      accY[r] += part[((wid * 8) + r) * 32 + lane];
#pragma unroll
    for (int r = 0; r < 8; ++r) {
      int t = i0 + r + mo;
      Y[((size_t)b * TT + t) * EE + h * DD + d0 + n] = (__bf16)accY[r];
    }
  }
}

// ---------------- out = Y @ Wo^T + bo (fp32 output) ----------------
__global__ void outproj_wmma(const __bf16* __restrict__ Yb, const __bf16* __restrict__ Wb,
                             const float* __restrict__ bias, float* __restrict__ out) {
  int lane = threadIdx.x & 31;
  int t0 = blockIdx.x * 16;
  int f0 = blockIdx.y * 16;
  int b  = blockIdx.z;
  const __bf16* Y = Yb + (size_t)b * TT * EE;
  v8f acc = {};
  for (int kk = 0; kk < EE; kk += 32) {
    v16bf a = load_a16(Y, t0 + (lane & 15), EE, kk, lane);
    v16bf w = load_b16_nk(Wb, f0 + (lane & 15), EE, kk, lane);
    acc = wmma_bf16(a, w, acc);
  }
  int n = lane & 15;
  int mo = (lane >> 4) * 8;
  int f = f0 + n;
  float bv = bias[f];
#pragma unroll
  for (int r = 0; r < 8; ++r) {
    int t = t0 + r + mo;
    out[((size_t)b * TT + t) * EE + f] = acc[r] + bv;
  }
}

extern "C" void kernel_launch(void* const* d_in, const int* in_sizes, int n_in,
                              void* d_out, int out_size, void* d_ws, size_t ws_size,
                              hipStream_t stream) {
  (void)in_sizes; (void)n_in; (void)out_size; (void)ws_size;
  const float* x   = (const float*)d_in[0];
  const float* emo = (const float*)d_in[1];
  const float* Wq  = (const float*)d_in[2];
  const float* bq  = (const float*)d_in[3];
  const float* Wk  = (const float*)d_in[4];
  const float* bk  = (const float*)d_in[5];
  const float* Wv  = (const float*)d_in[6];
  const float* bv  = (const float*)d_in[7];
  const float* Wo  = (const float*)d_in[8];
  const float* bo  = (const float*)d_in[9];
  const float* s   = (const float*)d_in[10];

  float* out  = (float*)d_out;
  float* attn = out + (size_t)NB * TT * EE;   // d_out = [out | attn]

  // workspace carve-out (~23 MB)
  char* w = (char*)d_ws;
  const size_t xbBytes = (size_t)NB * TT * EE * sizeof(__bf16);
  const size_t wBytes  = (size_t)EE * EE * sizeof(__bf16);
  __bf16* xb  = (__bf16*)w; w += xbBytes;
  __bf16* wqb = (__bf16*)w; w += wBytes;
  __bf16* wkb = (__bf16*)w; w += wBytes;
  __bf16* wvb = (__bf16*)w; w += wBytes;
  __bf16* wob = (__bf16*)w; w += wBytes;
  __bf16* Qb  = (__bf16*)w; w += xbBytes;   // [B,H,T,D]
  __bf16* Kb  = (__bf16*)w; w += xbBytes;   // [B,H,T,D]
  __bf16* Vtb = (__bf16*)w; w += xbBytes;   // [B,H,D,T] (transposed)
  __bf16* Yb  = (__bf16*)w; w += xbBytes;   // [B,T,E]
  float* stats = (float*)w;                 // NB * {mean, std}

  const int nx = NB * TT * EE;
  const int nw = EE * EE;
  cvt_bf16<<<(nx + 255) / 256, 256, 0, stream>>>(x, xb, nx);
  cvt_bf16<<<(nw + 255) / 256, 256, 0, stream>>>(Wq, wqb, nw);
  cvt_bf16<<<(nw + 255) / 256, 256, 0, stream>>>(Wk, wkb, nw);
  cvt_bf16<<<(nw + 255) / 256, 256, 0, stream>>>(Wv, wvb, nw);
  cvt_bf16<<<(nw + 255) / 256, 256, 0, stream>>>(Wo, wob, nw);

  bias_stats<<<NB, 256, 0, stream>>>(emo, stats);

  dim3 gproj(TT / 16, EE / 16, NB);
  proj_wmma<<<gproj, 32, 0, stream>>>(xb, wqb, bq, Qb, 0);
  proj_wmma<<<gproj, 32, 0, stream>>>(xb, wkb, bk, Kb, 0);
  proj_wmma<<<gproj, 32, 0, stream>>>(xb, wvb, bv, Vtb, 1);

  dim3 gattn(TT / 16, NB * NH);
  attn_fused<<<gattn, 256, LDS_BYTES, stream>>>(Qb, Kb, Vtb, emo, stats, s, attn, Yb);

  dim3 gout(TT / 16, EE / 16, NB);
  outproj_wmma<<<gout, 32, 0, stream>>>(Yb, wob, bo, out);
}
